// ResGatedDiGraphNet_2241972928895
// MI455X (gfx1250) — matrix operand order repaired and checked by reading, not compiled, not run on hardware
//
#include <hip/hip_runtime.h>

typedef __attribute__((ext_vector_type(16))) _Float16 v16h;
typedef __attribute__((ext_vector_type(8)))  _Float16 v8h;
typedef __attribute__((ext_vector_type(8)))  float    v8f;

#define WMMA_F16(a,b,c) __builtin_amdgcn_wmma_f32_16x16x32_f16(false,(a),false,(b),(short)0,(c),false,false)

// ---- WMMA operand fills (ISA 7.12.2 layouts, wave32) ------------------------

// A: 16x32 tile (MxK) from f32 row-major, converted to f16. Explicit b128 loads.
// lanes 0-15: row=lane, K = 0..7 (elems 0-7) and 16..23 (elems 8-15)
// lanes 16-31: row=lane-16, K = 8..15 and 24..31
__device__ __forceinline__ v16h load_a_tile(const float* __restrict__ base, int ld, int lane) {
  const int row = lane & 15;
  const int k0  = (lane & 16) ? 8 : 0;
  const float* p = base + row * ld + k0;
  const float4 f0 = *(const float4*)(p);
  const float4 f1 = *(const float4*)(p + 4);
  const float4 f2 = *(const float4*)(p + 16);
  const float4 f3 = *(const float4*)(p + 20);
  v16h a;
  a[0]  = (_Float16)f0.x; a[1]  = (_Float16)f0.y; a[2]  = (_Float16)f0.z; a[3]  = (_Float16)f0.w;
  a[4]  = (_Float16)f1.x; a[5]  = (_Float16)f1.y; a[6]  = (_Float16)f1.z; a[7]  = (_Float16)f1.w;
  a[8]  = (_Float16)f2.x; a[9]  = (_Float16)f2.y; a[10] = (_Float16)f2.z; a[11] = (_Float16)f2.w;
  a[12] = (_Float16)f3.x; a[13] = (_Float16)f3.y; a[14] = (_Float16)f3.z; a[15] = (_Float16)f3.w;
  return a;
}

// B: 32x16 tile (KxN) from pre-transposed f16 weights Wt[n][k] (64x64 per matrix).
// lanes 0-15: col=lane, K=0..7,16..23 ; lanes 16-31: col=lane-16, K=8..15,24..31.
// Per lane: two 8-f16 runs 16 apart = two 16B vector loads (global_load_b128).
__device__ __forceinline__ v16h load_bt_tile(const _Float16* __restrict__ Wt,
                                             int n_off, int k_off, int lane) {
  const int col = lane & 15;
  const int kb  = (lane & 16) ? 8 : 0;
  const _Float16* p = Wt + (n_off + col) * 64 + k_off + kb;
  const v8h lo = *(const v8h*)(p);
  const v8h hi = *(const v8h*)(p + 16);
  v16h b;
#pragma unroll
  for (int j = 0; j < 8; ++j) { b[j] = lo[j]; b[8 + j] = hi[j]; }
  return b;
}

// ---- Kernel 0: weight prep: f32 row-major [K=64][N=64] -> f16 transposed ----
// out[m][n*64+k] = (f16) in[m][k*64+n]; nmat matrices, coalesced writes.
__global__ __launch_bounds__(256) void wprep_kernel(const float* __restrict__ in,
                                                    _Float16* __restrict__ out, int nmat) {
  const int gid = blockIdx.x * 256 + threadIdx.x;
  if (gid >= nmat * 4096) return;
  const int m   = gid >> 12;
  const int idx = gid & 4095;
  const int n   = idx >> 6;
  const int k   = idx & 63;
  out[gid] = (_Float16)in[(m << 12) + (k << 6) + n];
}

// ---- Kernel 1: h0 = relu(x@W1+b1)@W2+b2  (fused, hidden tile via LDS) ------
__global__ __launch_bounds__(256) void mlp_kernel(
    const float* __restrict__ x,
    const _Float16* __restrict__ W1t, const float* __restrict__ b1,
    const _Float16* __restrict__ W2t, const float* __restrict__ b2,
    float* __restrict__ h0, int nRowTiles) {
  __shared__ float T[8][16 * 64];
  const int tid = threadIdx.x, wave = tid >> 5, lane = tid & 31;
  const int rt = blockIdx.x * 8 + wave;
  if (rt >= nRowTiles) return;                    // wave-uniform: EXEC stays all-ones
  const float* xt = x + (size_t)rt * (16 * 64);
  float* t = T[wave];
  const int col = lane & 15, r0 = (lane & 16) ? 8 : 0;

  v16h a0 = load_a_tile(xt,      64, lane);
  v16h a1 = load_a_tile(xt + 32, 64, lane);
#pragma unroll
  for (int nt = 0; nt < 4; ++nt) {
    const int n = nt * 16;
    v16h bt0 = load_bt_tile(W1t, n, 0,  lane);
    v16h bt1 = load_bt_tile(W1t, n, 32, lane);
    v8f c = {};
    c = WMMA_F16(a0, bt0, c);
    c = WMMA_F16(a1, bt1, c);
    const float bv = b1[n + col];
#pragma unroll
    for (int r = 0; r < 8; ++r) {
      float v = c[r] + bv;
      t[(r0 + r) * 64 + n + col] = v > 0.f ? v : 0.f;   // relu -> LDS (D-layout)
    }
  }
  // reload hidden tile from LDS in A-layout (cross-lane reshape via LDS)
  v16h h0a = load_a_tile(t,      64, lane);
  v16h h1a = load_a_tile(t + 32, 64, lane);
  float* ot = h0 + (size_t)rt * (16 * 64);
#pragma unroll
  for (int nt = 0; nt < 4; ++nt) {
    const int n = nt * 16;
    v16h bt0 = load_bt_tile(W2t, n, 0,  lane);
    v16h bt1 = load_bt_tile(W2t, n, 32, lane);
    v8f c = {};
    c = WMMA_F16(h0a, bt0, c);
    c = WMMA_F16(h1a, bt1, c);
    const float bv = b2[n + col];
#pragma unroll
    for (int r = 0; r < 8; ++r) ot[(r0 + r) * 64 + n + col] = c[r] + bv;
  }
}

// ---- Kernel 2: k|q|v -> kqv[N,192], self-term -> hout[N,64] -----------------
__global__ __launch_bounds__(256) void proj_kernel(
    const float* __restrict__ hin,
    const _Float16* __restrict__ Wkt, const float* __restrict__ bk,
    const _Float16* __restrict__ Wqt, const float* __restrict__ bq,
    const _Float16* __restrict__ Wvt, const float* __restrict__ bv,
    const _Float16* __restrict__ Wst, const float* __restrict__ bs,
    float* __restrict__ kqv, float* __restrict__ hout) {
  const int tid = threadIdx.x, wave = tid >> 5, lane = tid & 31;
  const int rt = blockIdx.x;
  const float* ht = hin + (size_t)rt * (16 * 64);
  const int col = lane & 15, r0 = (lane & 16) ? 8 : 0;

  v16h a0 = load_a_tile(ht,      64, lane);     // A reused across both col-tiles
  v16h a1 = load_a_tile(ht + 32, 64, lane);
#pragma unroll
  for (int it = 0; it < 2; ++it) {
    const int t    = wave + it * 8;             // 16 output col-tiles / 8 waves
    const int proj = t >> 2;                    // 0:k 1:q 2:v 3:self
    const int n    = (t & 3) * 16;
    const _Float16* Wt; const float* bb; float* obase; int ldo;
    if (proj == 0)      { Wt = Wkt; bb = bk; obase = kqv  + (size_t)rt * 16 * 192 + n;       ldo = 192; }
    else if (proj == 1) { Wt = Wqt; bb = bq; obase = kqv  + (size_t)rt * 16 * 192 + 64 + n;  ldo = 192; }
    else if (proj == 2) { Wt = Wvt; bb = bv; obase = kqv  + (size_t)rt * 16 * 192 + 128 + n; ldo = 192; }
    else                { Wt = Wst; bb = bs; obase = hout + (size_t)rt * 16 * 64 + n;        ldo = 64;  }
    v16h bt0 = load_bt_tile(Wt, n, 0,  lane);
    v16h bt1 = load_bt_tile(Wt, n, 32, lane);
    v8f c = {};
    c = WMMA_F16(a0, bt0, c);
    c = WMMA_F16(a1, bt1, c);
    const float bvv = bb[n + col];
#pragma unroll
    for (int r = 0; r < 8; ++r) obase[(r0 + r) * ldo + col] = c[r] + bvv;
  }
}

// ---- Kernel 3: per-edge gate + scatter-add (L2-resident, f32 HW atomics) ----
__global__ __launch_bounds__(256) void edge_kernel(
    const int* __restrict__ sA, const int* __restrict__ dA,
    const float* __restrict__ kqv, float* __restrict__ hout, int E) {
  const int gid = blockIdx.x * 256 + threadIdx.x;
  const int e = gid >> 4;
  if (e >= E) return;
  const int c = (gid & 15) << 2;                // 4 features per thread
  const int s = sA[e], d = dA[e];
  const float4 kk = *(const float4*)(kqv + (size_t)d * 192 + c);
  const float4 qq = *(const float4*)(kqv + (size_t)s * 192 + 64 + c);
  const float4 vv = *(const float4*)(kqv + (size_t)s * 192 + 128 + c);
  float* o = hout + (size_t)d * 64 + c;
  float g;
  g = 1.f / (1.f + __expf(-(kk.x + qq.x)));
  __hip_atomic_fetch_add(o + 0, g * vv.x, __ATOMIC_RELAXED, __HIP_MEMORY_SCOPE_AGENT);
  g = 1.f / (1.f + __expf(-(kk.y + qq.y)));
  __hip_atomic_fetch_add(o + 1, g * vv.y, __ATOMIC_RELAXED, __HIP_MEMORY_SCOPE_AGENT);
  g = 1.f / (1.f + __expf(-(kk.z + qq.z)));
  __hip_atomic_fetch_add(o + 2, g * vv.z, __ATOMIC_RELAXED, __HIP_MEMORY_SCOPE_AGENT);
  g = 1.f / (1.f + __expf(-(kk.w + qq.w)));
  __hip_atomic_fetch_add(o + 3, g * vv.w, __ATOMIC_RELAXED, __HIP_MEMORY_SCOPE_AGENT);
}

// ---- Kernel 4: score = concat(hfw,hbw) @ Wsc + bsc  (wave per node) ---------
__global__ __launch_bounds__(256) void score_kernel(
    const float* __restrict__ hfw, const float* __restrict__ hbw,
    const float* __restrict__ Wsc, const float* __restrict__ bsc,
    float* __restrict__ out, int N) {
  const int wave = threadIdx.x >> 5, lane = threadIdx.x & 31;
  const int n = blockIdx.x * 8 + wave;
  if (n >= N) return;
  const float* f = hfw + (size_t)n * 64;
  const float* b = hbw + (size_t)n * 64;
  float acc = f[lane] * Wsc[lane] + f[lane + 32] * Wsc[lane + 32]
            + b[lane] * Wsc[64 + lane] + b[lane + 32] * Wsc[96 + lane];
#pragma unroll
  for (int off = 16; off; off >>= 1) acc += __shfl_xor(acc, off, 32);
  if (lane == 0) out[n] = acc + bsc[0];
}

// ---- Host orchestration -----------------------------------------------------
extern "C" void kernel_launch(void* const* d_in, const int* in_sizes, int n_in,
                              void* d_out, int out_size, void* d_ws, size_t ws_size,
                              hipStream_t stream) {
  (void)n_in; (void)out_size; (void)ws_size;
  const float* x   = (const float*)d_in[0];
  const int*   ei  = (const int*)  d_in[1];
  const float* W1  = (const float*)d_in[2];
  const float* b1  = (const float*)d_in[3];
  const float* W2  = (const float*)d_in[4];
  const float* b2  = (const float*)d_in[5];
  const float* gWk = (const float*)d_in[6];
  const float* gbk = (const float*)d_in[7];
  const float* gWq = (const float*)d_in[8];
  const float* gbq = (const float*)d_in[9];
  const float* gWv = (const float*)d_in[10];
  const float* gbv = (const float*)d_in[11];
  const float* gWs = (const float*)d_in[12];
  const float* gb  = (const float*)d_in[13];
  const float* Wsc = (const float*)d_in[14];
  const float* bsc = (const float*)d_in[15];

  const int H = 64, L = 4;
  const int N = in_sizes[0] / H;          // 100000 (divisible by 16)
  const int E = in_sizes[1] / 2;          // 1200000
  const int rowTiles = N / 16;

  float* ws = (float*)d_ws;
  const size_t nh = (size_t)N * H;
  float* h0  = ws;                         // post-MLP features (both chains start here)
  float* bA  = ws + 1 * nh;
  float* bB  = ws + 2 * nh;
  float* bC  = ws + 3 * nh;
  float* kqv = ws + 4 * nh;                // N x 192 interleaved k|q|v
  _Float16* wt = (_Float16*)(ws + 7 * nh); // f16 transposed weights: [2 + 4*8] x 4096

  // one-shot weight prep (f32 -> f16 transposed, L2-resident thereafter)
  _Float16* W1t = wt;
  _Float16* W2t = wt + 4096;
  _Float16* Kt  = wt + 2 * 4096;           // 8 matrices (dir*4+l)
  _Float16* Qt  = wt + (2 + 8) * 4096;
  _Float16* Vt  = wt + (2 + 16) * 4096;
  _Float16* St  = wt + (2 + 24) * 4096;
  wprep_kernel<<<16,  256, 0, stream>>>(W1,  W1t, 1);
  wprep_kernel<<<16,  256, 0, stream>>>(W2,  W2t, 1);
  wprep_kernel<<<128, 256, 0, stream>>>(gWk, Kt,  8);
  wprep_kernel<<<128, 256, 0, stream>>>(gWq, Qt,  8);
  wprep_kernel<<<128, 256, 0, stream>>>(gWv, Vt,  8);
  wprep_kernel<<<128, 256, 0, stream>>>(gWs, St,  8);

  mlp_kernel<<<(rowTiles + 7) / 8, 256, 0, stream>>>(x, W1t, b1, W2t, b2, h0, rowTiles);

  const int egrid = (E * 16 + 255) / 256;
  const int* srcA = ei;        // edge_index[0]
  const int* dstA = ei + E;    // edge_index[1]

  // forward direction (dir 0): k/out indexed by dst, q/v by src
  const float* cur = h0;
  float* outsF[4] = {bA, bB, bA, bB};
  for (int l = 0; l < L; ++l) {
    const size_t mOff = (size_t)(0 * L + l) * 4096;
    const size_t bOff = (size_t)(0 * L + l) * H;
    proj_kernel<<<rowTiles, 256, 0, stream>>>(cur,
        Kt + mOff, gbk + bOff, Qt + mOff, gbq + bOff,
        Vt + mOff, gbv + bOff, St + mOff, gb + bOff, kqv, outsF[l]);
    edge_kernel<<<egrid, 256, 0, stream>>>(srcA, dstA, kqv, outsF[l], E);
    cur = outsF[l];
  }
  const float* hfw = cur;                  // bB

  // backward direction (dir 1): roles swapped (aggregate at original src)
  cur = h0;
  float* outsBw[4] = {bC, bA, bC, bA};     // bA is free again; bB holds hfw
  for (int l = 0; l < L; ++l) {
    const size_t mOff = (size_t)(1 * L + l) * 4096;
    const size_t bOff = (size_t)(1 * L + l) * H;
    proj_kernel<<<rowTiles, 256, 0, stream>>>(cur,
        Kt + mOff, gbk + bOff, Qt + mOff, gbq + bOff,
        Vt + mOff, gbv + bOff, St + mOff, gb + bOff, kqv, outsBw[l]);
    edge_kernel<<<egrid, 256, 0, stream>>>(dstA, srcA, kqv, outsBw[l], E);
    cur = outsBw[l];
  }
  const float* hbw = cur;                  // bA

  score_kernel<<<(N + 7) / 8, 256, 0, stream>>>(hfw, hbw, Wsc, bsc, (float*)d_out, N);
}
